// Masked_MAB_29892972380750
// MI455X (gfx1250) — matrix operand order repaired and checked by reading, not compiled, not run on hardware
//
#include <hip/hip_runtime.h>
#include <hip/hip_bf16.h>
#include <stdint.h>

// Masked MAB: B=8, N=1024, D=512, H=8, dh=64
#define NROWS  8192   // B*N
#define DMODEL 512
#define NH     8
#define DH     64
#define SEQ    1024
#define NB     8

typedef __attribute__((ext_vector_type(16))) _Float16 v16h;
typedef __attribute__((ext_vector_type(8)))  float    v8f;

union HFrag { v16h h; uint32_t u[8]; };

// A-matrix 16x32 f16 fragment (ISA 7.12.2): lane<16 holds K {0..7,16..23}, lane>=16 holds K {8..15,24..31}
__device__ __forceinline__ void fragA_from_row(HFrag& f, const _Float16* rowPtr, int half) {
  const uint32_t* p = (const uint32_t*)rowPtr + 4 * half;
#pragma unroll
  for (int j = 0; j < 4; ++j) { f.u[j] = p[j]; f.u[4 + j] = p[8 + j]; }
}

// B-matrix 32x16 f16 fragment: lane<16 holds K {0..7,16..23}, lane>=16 holds K {8..15,24..31}
__device__ __forceinline__ void fragB_from_row(HFrag& f, const _Float16* rowPtr, int half) {
  const uint32_t* p = (const uint32_t*)rowPtr + 4 * half;
#pragma unroll
  for (int j = 0; j < 4; ++j) { f.u[j] = p[j]; f.u[4 + j] = p[8 + j]; }
}

// ---------------------------------------------------------------------------
// GEMM: out = f((A @ W + bias) * mask)  ; A: [8192,512] f32, W: [512,512] f32
// MODE 0: outF = v             (Qp, f32)
// MODE 1: outH = (f16)v        (Kp / Vp)
// MODE 2: outF = res + relu(v) (final output)
// Block: 256 threads (8 waves), macro tile 128(M) x 64(N), K-step 32.
// Wave tile: 16 x 64 -> 4 WMMAs per K-step.
// ---------------------------------------------------------------------------
template <int MODE>
__global__ __launch_bounds__(256) void mab_gemm_kernel(
    const float* __restrict__ A, const float* __restrict__ W,
    const float* __restrict__ bias, const float* __restrict__ mask,
    const float* __restrict__ res,
    float* __restrict__ outF, _Float16* __restrict__ outH)
{
  __shared__ __attribute__((aligned(16))) _Float16 As[128 * 32];  // [m][k]
  __shared__ __attribute__((aligned(16))) _Float16 Bs[64 * 32];   // [n][k]

  const int t    = threadIdx.x;
  const int wave = t >> 5;
  const int lane = t & 31;
  const int l16  = lane & 15;
  const int half = lane >> 4;
  const int row0 = blockIdx.x * 128;
  const int col0 = blockIdx.y * 64;

  v8f acc[4] = {};
  for (int k0 = 0; k0 < DMODEL; k0 += 32) {
    __syncthreads();
    { // stage A tile 128x32, f32 -> f16
      const int m  = t >> 1;
      const int ks = (t & 1) * 16;
      const float4* src = (const float4*)(A + (size_t)(row0 + m) * DMODEL + k0 + ks);
      _Float16* dst = As + m * 32 + ks;
#pragma unroll
      for (int i = 0; i < 4; ++i) {
        float4 v = src[i];
        dst[i * 4 + 0] = (_Float16)v.x; dst[i * 4 + 1] = (_Float16)v.y;
        dst[i * 4 + 2] = (_Float16)v.z; dst[i * 4 + 3] = (_Float16)v.w;
      }
    }
    { // stage W tile transposed: Bs[n][k] = W[k0+k][col0+n]  (64x32)
      const int n  = t >> 2;
      const int ks = (t & 3) * 8;
      const float* src = W + (size_t)(k0 + ks) * DMODEL + col0 + n;
      _Float16* dst = Bs + n * 32 + ks;
#pragma unroll
      for (int i = 0; i < 8; ++i) dst[i] = (_Float16)src[(size_t)i * DMODEL];
    }
    __syncthreads();
    HFrag a;
    fragA_from_row(a, As + (wave * 16 + l16) * 32, half);
    HFrag b[4];
#pragma unroll
    for (int nn = 0; nn < 4; ++nn)
      fragB_from_row(b[nn], Bs + (nn * 16 + l16) * 32, half);
#pragma unroll
    for (int nn = 0; nn < 4; ++nn)
      acc[nn] = __builtin_amdgcn_wmma_f32_16x16x32_f16(false, a.h, false, b[nn].h,
                                                       (short)0, acc[nn], false, false);
  }

  float bc[4];
#pragma unroll
  for (int nn = 0; nn < 4; ++nn) bc[nn] = bias[col0 + nn * 16 + l16];
#pragma unroll
  for (int i = 0; i < 8; ++i) {
    const int row = row0 + wave * 16 + i + 8 * half;
    const float mk = mask[row];
    const size_t rowb = (size_t)row * DMODEL;
#pragma unroll
    for (int nn = 0; nn < 4; ++nn) {
      const float v = (acc[nn][i] + bc[nn]) * mk;
      const size_t idx = rowb + col0 + nn * 16 + l16;
      if (MODE == 0) {
        outF[idx] = v;
      } else if (MODE == 1) {
        outH[idx] = (_Float16)v;
      } else {
        const float r = v > 0.f ? v : 0.f;
        outF[idx] = res[idx] + r;
      }
    }
  }
}

// ---------------------------------------------------------------------------
// Flash attention per (h,b): O = q + softmax_masked(q k^T / sqrt(512)) v
// Block = 8 waves, all on same (h,b); wave owns 16 query rows x dh=64.
// Key tiles of 32, online softmax via LDS round-trip.
// K tile staged with CDNA5 async global->LDS copies (ASYNCcnt).
// ---------------------------------------------------------------------------
__global__ __launch_bounds__(256) void mab_attn_kernel(
    const float* __restrict__ Qp, const _Float16* __restrict__ Kp,
    const _Float16* __restrict__ Vp, const float* __restrict__ mask,
    float* __restrict__ Obuf)
{
  __shared__ __attribute__((aligned(16))) _Float16 Ks[32 * 64];       // [key][d]
  __shared__ __attribute__((aligned(16))) _Float16 Vt[64 * 32];       // [d][key]
  __shared__ float    maskk[32];
  __shared__ float    Sbuf[8][16 * 32];                               // per-wave S
  __shared__ __attribute__((aligned(16))) _Float16 Pbuf[8][16 * 32];  // per-wave P
  __shared__ float    rowscale[8][16];
  __shared__ float    rowsum[8][16];

  const int t    = threadIdx.x;
  const int wave = t >> 5;
  const int lane = t & 31;
  const int l16  = lane & 15;
  const int half = lane >> 4;

  const int hb    = blockIdx.x;
  const int h     = hb & (NH - 1);
  const int bb    = hb >> 3;
  const int hoff  = h * DH;
  const int rbase = bb * SEQ;
  const int n0    = blockIdx.y * 128 + wave * 16;

  // q fragments (A layout), loaded once from f32 Qp
  HFrag qf0, qf1;
  {
    const int grow = rbase + n0 + l16;
    const float* qrow = Qp + (size_t)grow * DMODEL + hoff;
#pragma unroll
    for (int j = 0; j < 8; ++j) {
      const int kp = (j < 4) ? (2 * j + 8 * half) : (16 + 2 * (j - 4) + 8 * half);
      qf0.h[2 * j]     = (_Float16)qrow[kp];
      qf0.h[2 * j + 1] = (_Float16)qrow[kp + 1];
      qf1.h[2 * j]     = (_Float16)qrow[32 + kp];
      qf1.h[2 * j + 1] = (_Float16)qrow[32 + kp + 1];
    }
  }

  v8f acc0 = {}, acc1 = {}, acc2 = {}, acc3 = {};
  float m_r = -3.0e38f, s_r = 0.f;
  const float scale = 0.044194173824159216f;  // 1/sqrt(512)

  for (int kt = 0; kt < SEQ / 32; ++kt) {
    const int kb = kt * 32;
    __syncthreads();
    { // stage K tile [32 keys][64 d]: async global->LDS B128, one per thread
      const int key = t >> 3;
      const int ds  = (t & 7) * 8;
      const _Float16* gsrc = Kp + (size_t)(rbase + kb + key) * DMODEL + hoff + ds;
      const unsigned lds_off = (unsigned)(uintptr_t)(Ks + key * 64 + ds);
      asm volatile("global_load_async_to_lds_b128 %0, %1, off"
                   :: "v"(lds_off), "v"(gsrc)
                   : "memory");
    }
    { // stage V transposed: Vt[d][key]
      const int key = t & 31;
      const int ds  = (t >> 5) * 8;
      const _Float16* src = Vp + (size_t)(rbase + kb + key) * DMODEL + hoff + ds;
#pragma unroll
      for (int i = 0; i < 8; ++i) Vt[(ds + i) * 32 + key] = src[i];
    }
    if (t < 32) maskk[t] = mask[rbase + kb + t];
    asm volatile("s_wait_asynccnt 0x0" ::: "memory");
    __syncthreads();

    // S = (q k^T) * scale : two 16-key sub-tiles, K=64 = 2 wmma each
#pragma unroll
    for (int nn = 0; nn < 2; ++nn) {
      HFrag b0, b1;
      const _Float16* krow = Ks + (nn * 16 + l16) * 64;
      fragB_from_row(b0, krow, half);
      fragB_from_row(b1, krow + 32, half);
      v8f s = {};
      s = __builtin_amdgcn_wmma_f32_16x16x32_f16(false, qf0.h, false, b0.h, (short)0, s, false, false);
      s = __builtin_amdgcn_wmma_f32_16x16x32_f16(false, qf1.h, false, b1.h, (short)0, s, false, false);
#pragma unroll
      for (int i = 0; i < 8; ++i)
        Sbuf[wave][(i + 8 * half) * 32 + nn * 16 + l16] = s[i] * scale;
    }
    __syncthreads();

    // online softmax; lanes 0..15 each own one query row
    if (lane < 16) {
      const int r = lane;
      float tmax = m_r;
#pragma unroll
      for (int j = 0; j < 32; ++j) {
        const float x = Sbuf[wave][r * 32 + j];
        if (maskk[j] > 0.f) tmax = fmaxf(tmax, x);
      }
      const float sc = __expf(m_r - tmax);
      float ssum = s_r * sc;
#pragma unroll
      for (int j = 0; j < 32; ++j) {
        const float x = Sbuf[wave][r * 32 + j];
        const float p = (maskk[j] > 0.f) ? __expf(x - tmax) : 0.f;
        ssum += p;
        Pbuf[wave][r * 32 + j] = (_Float16)p;
      }
      m_r = tmax; s_r = ssum;
      rowscale[wave][r] = sc;
    }
    __syncthreads();

    // rescale accumulators, then O += P @ V (4 d-subtiles)
    float rsc[8];
#pragma unroll
    for (int i = 0; i < 8; ++i) rsc[i] = rowscale[wave][i + 8 * half];
#pragma unroll
    for (int i = 0; i < 8; ++i) {
      acc0[i] *= rsc[i]; acc1[i] *= rsc[i]; acc2[i] *= rsc[i]; acc3[i] *= rsc[i];
    }
    HFrag p;
    fragA_from_row(p, Pbuf[wave] + l16 * 32, half);
    HFrag v0, v1, v2, v3;
    fragB_from_row(v0, Vt + (0 * 16 + l16) * 32, half);
    fragB_from_row(v1, Vt + (1 * 16 + l16) * 32, half);
    fragB_from_row(v2, Vt + (2 * 16 + l16) * 32, half);
    fragB_from_row(v3, Vt + (3 * 16 + l16) * 32, half);
    acc0 = __builtin_amdgcn_wmma_f32_16x16x32_f16(false, p.h, false, v0.h, (short)0, acc0, false, false);
    acc1 = __builtin_amdgcn_wmma_f32_16x16x32_f16(false, p.h, false, v1.h, (short)0, acc1, false, false);
    acc2 = __builtin_amdgcn_wmma_f32_16x16x32_f16(false, p.h, false, v2.h, (short)0, acc2, false, false);
    acc3 = __builtin_amdgcn_wmma_f32_16x16x32_f16(false, p.h, false, v3.h, (short)0, acc3, false, false);
  }

  __syncthreads();
  if (lane < 16) rowsum[wave][lane] = s_r;
  __syncthreads();

#pragma unroll
  for (int i = 0; i < 8; ++i) {
    const int r    = i + 8 * half;
    const int grow = rbase + n0 + r;
    const float inv = mask[grow] / (rowsum[wave][r] + 1e-16f);
    const size_t base = (size_t)grow * DMODEL + hoff;
    Obuf[base + 0 * 16 + l16] = Qp[base + 0 * 16 + l16] + acc0[i] * inv;
    Obuf[base + 1 * 16 + l16] = Qp[base + 1 * 16 + l16] + acc1[i] * inv;
    Obuf[base + 2 * 16 + l16] = Qp[base + 2 * 16 + l16] + acc2[i] * inv;
    Obuf[base + 3 * 16 + l16] = Qp[base + 3 * 16 + l16] + acc3[i] * inv;
  }
}

extern "C" void kernel_launch(void* const* d_in, const int* in_sizes, int n_in,
                              void* d_out, int out_size, void* d_ws, size_t ws_size,
                              hipStream_t stream) {
  (void)in_sizes; (void)n_in; (void)out_size; (void)ws_size;
  const float* Q    = (const float*)d_in[0];
  const float* K    = (const float*)d_in[1];
  const float* mask = (const float*)d_in[2];
  const float* Wq   = (const float*)d_in[3];
  const float* bq   = (const float*)d_in[4];
  const float* Wk   = (const float*)d_in[5];
  const float* bk   = (const float*)d_in[6];
  const float* Wv   = (const float*)d_in[7];
  const float* bv   = (const float*)d_in[8];
  const float* Wo   = (const float*)d_in[9];
  const float* bo   = (const float*)d_in[10];
  float* out = (float*)d_out;

  // Workspace layout (48 MB total):
  char* ws = (char*)d_ws;
  float*    Qp   = (float*)ws;                                  // 16 MB f32
  float*    Obuf = (float*)(ws + (size_t)16 * 1024 * 1024);     // 16 MB f32
  _Float16* Kp   = (_Float16*)(ws + (size_t)32 * 1024 * 1024);  //  8 MB f16
  _Float16* Vp   = (_Float16*)(ws + (size_t)40 * 1024 * 1024);  //  8 MB f16

  dim3 gg(NROWS / 128, DMODEL / 64);
  mab_gemm_kernel<0><<<gg, 256, 0, stream>>>(Q, Wq, bq, mask, nullptr, Qp, nullptr);
  mab_gemm_kernel<1><<<gg, 256, 0, stream>>>(K, Wk, bk, mask, nullptr, nullptr, Kp);
  mab_gemm_kernel<1><<<gg, 256, 0, stream>>>(K, Wv, bv, mask, nullptr, nullptr, Vp);
  mab_attn_kernel<<<dim3(NB * NH, SEQ / 128), 256, 0, stream>>>(Qp, Kp, Vp, mask, Obuf);
  mab_gemm_kernel<2><<<gg, 256, 0, stream>>>(Obuf, Wo, bo, mask, Obuf, out, nullptr);
}
